// WindowAttention_70901320122611
// MI455X (gfx1250) — matrix operand order repaired and checked by reading, not compiled, not run on hardware
//
#include <hip/hip_runtime.h>
#include <hip/hip_bf16.h>

// ---------------------------------------------------------------------------
// Shifted-window attention (Swin block) for MI455X / gfx1250, wave32 + WMMA.
// Pipeline: stage(f32->f16, roll, window order) -> QKV GEMM -> windowed
// attention (QK^T, bias+mask, softmax, PV) -> output projection (+bias, roll).
// All matmuls use v_wmma_f32_16x16x32_f16.
// ---------------------------------------------------------------------------

typedef _Float16 h16 __attribute__((ext_vector_type(16)));
typedef _Float16 h8  __attribute__((ext_vector_type(8)));
typedef float    f8  __attribute__((ext_vector_type(8)));

union V16 { h16 v; h8 h[2]; };

#define WSZ      8
#define HEADS    16
#define HEAD_DIM 32
#define DIM      512
#define NTOK     64                 // tokens per window (8x8)
#define NWIN     64                 // windows per image (8x8)
#define BATCH    16
#define TOKENS   (BATCH * NWIN * NTOK)   // 65536

// ---------------------------------------------------------------------------
// Kernel 1: convert weights to f16 (w_qkv [1536,512], w_out [512,512])
// ---------------------------------------------------------------------------
__global__ void swin_convert_weights(const float* __restrict__ wqkv,
                                     const float* __restrict__ wout,
                                     _Float16* __restrict__ wqkvh,
                                     _Float16* __restrict__ wouth) {
    int i = blockIdx.x * blockDim.x + threadIdx.x;
    if (i < 3 * HEADS * HEAD_DIM * DIM) wqkvh[i] = (_Float16)wqkv[i];
    if (i < DIM * DIM)                  wouth[i] = (_Float16)wout[i];
}

// ---------------------------------------------------------------------------
// Kernel 2: stage x -> f16, fusing roll(-4,-4) and window-token ordering.
// Output xh: [token m][512] with m = ((b*64 + win)*64 + ti).
// ---------------------------------------------------------------------------
__global__ void swin_stage_x(const float* __restrict__ x,
                             _Float16* __restrict__ xh) {
    int m   = blockIdx.x;          // 0..65535
    int bi  = m >> 12;
    int rem = m & 4095;
    int win = rem >> 6, ti = rem & 63;
    int sy = (((win >> 3) << 3) + (ti >> 3) + 4) & 63;   // rolled source row
    int sx = (((win & 7) << 3) + (ti & 7) + 4) & 63;     // rolled source col
    const float* src = x + (((size_t)bi * 64 + sy) * 64 + sx) * DIM;
    _Float16*    dst = xh + (size_t)m * DIM;
    int c = threadIdx.x * 4;                              // 128 thr * 4 = 512
    float4 f = *(const float4*)(src + c);
    dst[c + 0] = (_Float16)f.x;
    dst[c + 1] = (_Float16)f.y;
    dst[c + 2] = (_Float16)f.z;
    dst[c + 3] = (_Float16)f.w;
}

// ---------------------------------------------------------------------------
// Kernel 3: QKV GEMM.  D[m][n] = sum_c xh[m][c] * wqkv[n][c]
// grid = (1536/64, 65536/64), block = 128 (4 waves, each 16 rows x 64 cols).
// Results scattered to q/k/v buffers laid out [b][head][win][tok][32] (f16).
// ---------------------------------------------------------------------------
__global__ void swin_qkv_gemm(const _Float16* __restrict__ xh,
                              const _Float16* __restrict__ wh,
                              _Float16* __restrict__ qb,
                              _Float16* __restrict__ kb,
                              _Float16* __restrict__ vb) {
    const int lane = threadIdx.x & 31;
    const int wave = threadIdx.x >> 5;
    const int l16  = (lane < 16) ? 0 : 1;
    const int ln   = lane & 15;
    const int mBase = blockIdx.y * 64 + wave * 16;
    const int nBase = blockIdx.x * 64;

    f8 acc[4] = {};
    const _Float16* aRow = xh + (size_t)(mBase + ln) * DIM;

    for (int k0 = 0; k0 < DIM; k0 += 32) {
        // A (16x32 f16): lane row = ln; lanes<16 hold K[0..7],[16..23],
        // lanes>=16 hold K[8..15],[24..31]  (ISA 7.12.2)
        V16 a;
        const _Float16* ap = aRow + k0 + l16 * 8;
        a.h[0] = *(const h8*)(ap);
        a.h[1] = *(const h8*)(ap + 16);
        __builtin_prefetch(aRow + k0 + 128, 0, 0);
#pragma unroll
        for (int nt = 0; nt < 4; ++nt) {
            // B (32x16 f16): lane column = ln; lanes<16 hold K[0..7],[16..23],
            // lanes>=16 hold K[8..15],[24..31] (same order as A)
            const _Float16* bp =
                wh + (size_t)(nBase + nt * 16 + ln) * DIM + k0 + l16 * 8;
            V16 b;
            b.h[0] = *(const h8*)(bp);
            b.h[1] = *(const h8*)(bp + 16);
            acc[nt] = __builtin_amdgcn_wmma_f32_16x16x32_f16(
                false, a.v, false, b.v, (short)0, acc[nt], false, false);
        }
    }
    // epilogue: n -> (s, head, d); scatter f16 into q/k/v
#pragma unroll
    for (int nt = 0; nt < 4; ++nt) {
        int n  = nBase + nt * 16 + ln;
        int s  = n >> 9;               // 0=q 1=k 2=v
        int hh = (n >> 5) & 15;
        int d  = n & 31;
        _Float16* buf = (s == 0) ? qb : (s == 1) ? kb : vb;
#pragma unroll
        for (int r = 0; r < 8; ++r) {
            int m   = mBase + r + l16 * 8;   // C layout: VGPR r -> M=r / r+8
            int bi  = m >> 12;
            int rem = m & 4095;
            int win = rem >> 6, ti = rem & 63;
            size_t idx =
                ((((size_t)bi * HEADS + hh) * NWIN + win) * NTOK + ti) *
                    HEAD_DIM + d;
            buf[idx] = (_Float16)acc[nt][r];
        }
    }
}

// ---------------------------------------------------------------------------
// Kernel 4: windowed attention, one block per (b, head, window).
// S = q k^T * scale + bias + shift-mask ; softmax ; O = P v.
// Output obuf: [b][win][tok][head*32 + d] (f16) for the projection GEMM.
// ---------------------------------------------------------------------------
__global__ void swin_attention(const _Float16* __restrict__ qbuf,
                               const _Float16* __restrict__ kbuf,
                               const _Float16* __restrict__ vbuf,
                               const float* __restrict__ pos,
                               _Float16* __restrict__ obuf) {
    __shared__ __align__(16) float    S[64 * 64];
    __shared__ __align__(32) _Float16 P[64 * 64];
    __shared__ __align__(32) _Float16 vT[32 * 64];
    __shared__ float posLds[225];

    const int t    = threadIdx.x;
    const int lane = t & 31, wave = t >> 5;
    const int l16  = (lane < 16) ? 0 : 1;
    const int ln   = lane & 15;
    const int win  = blockIdx.x & 63;
    const int hh   = (blockIdx.x >> 6) & 15;
    const int bi   = blockIdx.x >> 10;
    const size_t base =
        (((size_t)bi * HEADS + hh) * NWIN + win) * (NTOK * HEAD_DIM);

    for (int i = t; i < 225; i += 128) posLds[i] = pos[i];
    // stage v transposed: vT[d][j] = v[j][d]  (contiguous global reads)
#pragma unroll
    for (int e = 0; e < 16; ++e) {
        int idx = t * 16 + e;
        int j = idx >> 5, d = idx & 31;
        vT[d * 64 + j] = vbuf[base + idx];
    }

    // ---- S = q @ k^T (K = 32, single WMMA step; 4 column tiles) ----
    f8 c[4];
    {
        V16 a;
        const _Float16* ap =
            qbuf + base + (size_t)(wave * 16 + ln) * HEAD_DIM + l16 * 8;
        a.h[0] = *(const h8*)(ap);
        a.h[1] = *(const h8*)(ap + 16);
#pragma unroll
        for (int jt = 0; jt < 4; ++jt) {
            // B = k^T: column j holds k[j][*]; same k order as A
            const _Float16* bp =
                kbuf + base + (size_t)(jt * 16 + ln) * HEAD_DIM + l16 * 8;
            V16 b;
            b.h[0] = *(const h8*)(bp);
            b.h[1] = *(const h8*)(bp + 16);
            f8 z = {};
            c[jt] = __builtin_amdgcn_wmma_f32_16x16x32_f16(
                false, a.v, false, b.v, (short)0, z, false, false);
        }
    }
    __syncthreads();   // posLds + vT ready

    const float scale  = 0.17677669529663687f;   // 32^-0.5
    const bool maskUL  = (win >> 3) == 7;        // last window-row
    const bool maskLR  = (win & 7) == 7;         // last window-col
#pragma unroll
    for (int jt = 0; jt < 4; ++jt) {
#pragma unroll
        for (int r = 0; r < 8; ++r) {
            int i  = wave * 16 + r + l16 * 8;
            int j  = jt * 16 + ln;
            int d0 = (j >> 3) - (i >> 3) + 7;
            int d1 = (j & 7) - (i & 7) + 7;
            float val = c[jt][r] * scale + posLds[d0 * 15 + d1];
            if (maskUL && ((i >= 32) != (j >= 32)))           val = -1e30f;
            if (maskLR && (((i & 7) >= 4) != ((j & 7) >= 4))) val = -1e30f;
            S[i * 64 + j] = val;
        }
    }
    __syncthreads();

    // ---- row softmax (64 rows, one thread each) ----
    if (t < 64) {
        float m = -1e30f;
        for (int j = 0; j < 64; ++j) m = fmaxf(m, S[t * 64 + j]);
        float sum = 0.f;
        for (int j = 0; j < 64; ++j) {
            float e = __expf(S[t * 64 + j] - m);
            S[t * 64 + j] = e;
            sum += e;
        }
        float inv = 1.0f / sum;
        for (int j = 0; j < 64; ++j)
            P[t * 64 + j] = (_Float16)(S[t * 64 + j] * inv);
    }
    __syncthreads();

    // ---- O = P @ v   (K = 64 -> two WMMA steps; N = 32 -> two tiles) ----
    f8 o[2] = {};
#pragma unroll
    for (int ks = 0; ks < 64; ks += 32) {
        V16 a;
        const _Float16* ap = &P[(wave * 16 + ln) * 64 + ks + l16 * 8];
        a.h[0] = *(const h8*)(ap);
        a.h[1] = *(const h8*)(ap + 16);
#pragma unroll
        for (int nt = 0; nt < 2; ++nt) {
            const _Float16* bp = &vT[(nt * 16 + ln) * 64 + ks + l16 * 8];
            V16 b;
            b.h[0] = *(const h8*)(bp);
                b.h[1] = *(const h8*)(bp + 16);
            o[nt] = __builtin_amdgcn_wmma_f32_16x16x32_f16(
                false, a.v, false, b.v, (short)0, o[nt], false, false);
        }
    }
#pragma unroll
    for (int nt = 0; nt < 2; ++nt) {
        int d = nt * 16 + ln;
#pragma unroll
        for (int r = 0; r < 8; ++r) {
            int i = wave * 16 + r + l16 * 8;
            size_t idx = (((size_t)bi * NWIN + win) * NTOK + i) * DIM +
                         hh * HEAD_DIM + d;
            obuf[idx] = (_Float16)o[nt][r];
        }
    }
}

// ---------------------------------------------------------------------------
// Kernel 5: output projection + bias, roll(+4,+4) fused into the store.
// grid = (512/64, 65536/64), block = 128.
// ---------------------------------------------------------------------------
__global__ void swin_proj(const _Float16* __restrict__ ah,
                          const _Float16* __restrict__ wh,
                          const float* __restrict__ bias,
                          float* __restrict__ out) {
    const int lane = threadIdx.x & 31;
    const int wave = threadIdx.x >> 5;
    const int l16  = (lane < 16) ? 0 : 1;
    const int ln   = lane & 15;
    const int mBase = blockIdx.y * 64 + wave * 16;
    const int nBase = blockIdx.x * 64;

    f8 acc[4] = {};
    const _Float16* aRow = ah + (size_t)(mBase + ln) * DIM;
    for (int k0 = 0; k0 < DIM; k0 += 32) {
        V16 a;
        const _Float16* ap = aRow + k0 + l16 * 8;
        a.h[0] = *(const h8*)(ap);
        a.h[1] = *(const h8*)(ap + 16);
        __builtin_prefetch(aRow + k0 + 128, 0, 0);
#pragma unroll
        for (int nt = 0; nt < 4; ++nt) {
            const _Float16* bp =
                wh + (size_t)(nBase + nt * 16 + ln) * DIM + k0 + l16 * 8;
            V16 b;
            b.h[0] = *(const h8*)(bp);
            b.h[1] = *(const h8*)(bp + 16);
            acc[nt] = __builtin_amdgcn_wmma_f32_16x16x32_f16(
                false, a.v, false, b.v, (short)0, acc[nt], false, false);
        }
    }
#pragma unroll
    for (int nt = 0; nt < 4; ++nt) {
        int n   = nBase + nt * 16 + ln;
        float bn = bias[n];
#pragma unroll
        for (int r = 0; r < 8; ++r) {
            int m   = mBase + r + l16 * 8;
            int bi  = m >> 12;
            int rem = m & 4095;
            int win = rem >> 6, ti = rem & 63;
            int y = (((win >> 3) << 3) + (ti >> 3) + 4) & 63;  // un-shift
            int x = (((win & 7) << 3) + (ti & 7) + 4) & 63;
            out[(((size_t)bi * 64 + y) * 64 + x) * DIM + n] = acc[nt][r] + bn;
        }
    }
}

// ---------------------------------------------------------------------------
// Host launcher
// ---------------------------------------------------------------------------
extern "C" void kernel_launch(void* const* d_in, const int* in_sizes, int n_in,
                              void* d_out, int out_size, void* d_ws,
                              size_t ws_size, hipStream_t stream) {
    const float* x      = (const float*)d_in[0];   // [16,64,64,512]
    const float* w_qkv  = (const float*)d_in[1];   // [1536,512]
    const float* pos    = (const float*)d_in[2];   // [15,15]
    const float* w_out  = (const float*)d_in[3];   // [512,512]
    const float* b_out  = (const float*)d_in[4];   // [512]
    float* out = (float*)d_out;

    // Workspace layout (bytes):
    //   [0, 64MiB)        xh staged input (f16), later reused as attn output
    //   [64MiB, 128MiB)   q   (f16)
    //   [128MiB, 192MiB)  k   (f16)
    //   [192MiB, 256MiB)  v   (f16)
    //   [256MiB, +1.5MiB) w_qkv f16
    //   then +0.5MiB      w_out f16
    char* ws = (char*)d_ws;
    _Float16* xh    = (_Float16*)(ws);                       // also attn out
    _Float16* qb    = (_Float16*)(ws + (size_t)64  * 1024 * 1024);
    _Float16* kb    = (_Float16*)(ws + (size_t)128 * 1024 * 1024);
    _Float16* vb    = (_Float16*)(ws + (size_t)192 * 1024 * 1024);
    _Float16* wqkvh = (_Float16*)(ws + (size_t)256 * 1024 * 1024);
    _Float16* wouth = (_Float16*)(ws + (size_t)256 * 1024 * 1024 + 1572864);

    // 1) weights -> f16
    swin_convert_weights<<<3072, 256, 0, stream>>>(w_qkv, w_out, wqkvh, wouth);
    // 2) x -> f16 with roll(-4,-4) + window ordering
    swin_stage_x<<<TOKENS, 128, 0, stream>>>(x, xh);
    // 3) QKV projection (WMMA)
    swin_qkv_gemm<<<dim3(24, TOKENS / 64), 128, 0, stream>>>(xh, wqkvh,
                                                             qb, kb, vb);
    // 4) windowed attention (WMMA x2 + softmax); writes into xh region
    swin_attention<<<BATCH * HEADS * NWIN, 128, 0, stream>>>(qb, kb, vb, pos,
                                                             xh);
    // 5) output projection + bias + roll(+4,+4)
    swin_proj<<<dim3(8, TOKENS / 64), 128, 0, stream>>>(xh, wouth, b_out, out);

    (void)in_sizes; (void)n_in; (void)out_size; (void)ws_size;
}
